// NestedParallelBlock_67937792688180
// MI455X (gfx1250) — compile-verified, matrix operand order checked by reading, not run on hardware
//
#include <hip/hip_runtime.h>
#include <hip/hip_bf16.h>
#include <math.h>

// ---------------------------------------------------------------- types
typedef __attribute__((ext_vector_type(16))) __bf16 v16bf;
typedef __attribute__((ext_vector_type(8)))  float  v8f;

#define DIMD   512
#define NHEAD  8
#define HD     64
#define EXPN   3584      // 3D + 4D
#define CATD   2560      // 4D + D
#define COND   1024      // 2D
#define MTOT   16384     // B*N tokens
#define QSCALE 0.125f    // (D/H)^-0.5 = 64^-0.5

// float -> bf16 round-to-nearest-even
__device__ __forceinline__ unsigned short f2bf(float f) {
  unsigned u = __builtin_bit_cast(unsigned, f);
  u = (u + 0x7FFFu + ((u >> 16) & 1u)) >> 16;
  return (unsigned short)u;
}
// fragment element (kh = lane>>4, e in [0,16)) -> k-local in [0,32)
__device__ __forceinline__ int inv_klocal(int kh, int e) {
  return (e & 7) + (kh << 3) + ((e >> 3) << 4);
}
__device__ __forceinline__ v8f wmma_bf16(v16bf a, v16bf b, v8f c) {
  return __builtin_amdgcn_wmma_f32_16x16x32_bf16(false, a, false, b,
                                                 (short)0, c, false, false);
}
__device__ __forceinline__ void lds_fence() {
  asm volatile("s_wait_dscnt 0" ::: "memory");
}
__device__ __forceinline__ void sched_fence() {
  __builtin_amdgcn_sched_barrier(0);   // keep prefetched loads above the WMMAs
}
// read 16 fragment elements from two contiguous 8-float runs, store 32B packed
__device__ __forceinline__ void pack16_store(const float* __restrict__ run0,
                                             const float* __restrict__ run1,
                                             unsigned short* __restrict__ dst) {
  union { unsigned short s[16]; uint4 q[2]; } u;
#pragma unroll
  for (int e = 0; e < 8; e++) u.s[e] = f2bf(run0[e]);
#pragma unroll
  for (int e = 0; e < 8; e++) u.s[8 + e] = f2bf(run1[e]);
  uint4* o4 = (uint4*)dst;
  o4[0] = u.q[0];
  o4[1] = u.q[1];
}

// ============================================================== weight pack
// Row-major W[N x K] (f32) -> B-fragment layout [N/16][K/32][32 lanes][16 bf16]
__global__ void pack_b_kernel(const float* __restrict__ W,
                              unsigned short* __restrict__ dst,
                              int ntiles, int ktiles, int K) {
  int t = blockIdx.x * blockDim.x + threadIdx.x;
  int total = ntiles * ktiles * 32;
  if (t >= total) return;
  int lane = t & 31;
  int kt   = (t >> 5) % ktiles;
  int nt   = (t >> 5) / ktiles;
  int n = lane & 15, kh = lane >> 4;
  const float* src = W + (size_t)(nt * 16 + n) * K + kt * 32;
  union { unsigned short s[16]; uint4 q[2]; } u;
#pragma unroll
  for (int e = 0; e < 16; e++) u.s[e] = f2bf(src[inv_klocal(kh, e)]);
  uint4* o4 = (uint4*)(dst + (size_t)t * 16);
  o4[0] = u.q[0];
  o4[1] = u.q[1];
}

// ============================================================== LN1 + pack h
// one wave per token: layernorm over 512, expert mask, A-frag pack (bf16)
__global__ __launch_bounds__(256) void ln1_pack_kernel(
    const float* __restrict__ x, const int* __restrict__ emask,
    const float* __restrict__ w, const float* __restrict__ b,
    unsigned short* __restrict__ hfrag) {
  int m    = (blockIdx.x * blockDim.x + threadIdx.x) >> 5;   // token
  int lane = threadIdx.x & 31;
  const float* row = x + (size_t)m * DIMD + lane * 16;       // 16 contiguous/lane
  float vals[16], s = 0.f, s2 = 0.f;
#pragma unroll
  for (int t4 = 0; t4 < 4; t4++) {
    float4 r = *(const float4*)(row + t4 * 4);
    vals[t4 * 4 + 0] = r.x; vals[t4 * 4 + 1] = r.y;
    vals[t4 * 4 + 2] = r.z; vals[t4 * 4 + 3] = r.w;
    s += r.x + r.y + r.z + r.w;
    s2 += r.x * r.x + r.y * r.y + r.z * r.z + r.w * r.w;
  }
#pragma unroll
  for (int o = 16; o >= 1; o >>= 1) {
    s  += __shfl_xor(s,  o, 32);
    s2 += __shfl_xor(s2, o, 32);
  }
  float mean = s * (1.f / DIMD);
  float rstd = rsqrtf(s2 * (1.f / DIMD) - mean * mean + 1e-5f);
  int dtok = DIMD >> emask[m];
  int mt = m >> 4, mr = m & 15;
#pragma unroll
  for (int t = 0; t < 16; t++) {
    int c = lane * 16 + t;
    float h = (c < dtok) ? ((vals[t] - mean) * rstd * w[c] + b[c]) : 0.f;
    int kt = c >> 5, kl = c & 31;
    int kh = (kl >> 3) & 1;
    int e  = (kl & 7) + ((kl >> 4) << 3);
    hfrag[(size_t)(((mt * 16 + kt) * 32) + (mr + 16 * kh)) * 16 + e] = f2bf(h);
  }
}

// ============================================================== expand GEMM
// Y[16384 x 3584] = h @ Wexp^T + bias; wave computes 16x64 tile.
// Distance-1 pipelined fragment stream pinned with sched_barrier.
__global__ __launch_bounds__(256) void gemm_expand_kernel(
    const unsigned short* __restrict__ hfrag,
    const unsigned short* __restrict__ wexpf,
    const float* __restrict__ mlp_bias,
    float* __restrict__ bufKV,
    unsigned short* __restrict__ qfrag,
    unsigned short* __restrict__ catfrag) {
  __shared__ float lds[8 * 1024];
  int wv = threadIdx.x >> 5, lane = threadIdx.x & 31;
  int w  = blockIdx.x * 8 + wv;
  int mt = w / 56, ng = w % 56;
  int n0 = ng * 4;                       // first of 4 N-tiles (of 224)
  const v16bf* A  = (const v16bf*)hfrag;
  const v16bf* Bw = (const v16bf*)wexpf;
  size_t aB = (size_t)(mt * 16) * 32 + lane;
  size_t bB0 = (size_t)((n0 + 0) * 16) * 32 + lane;
  size_t bB1 = (size_t)((n0 + 1) * 16) * 32 + lane;
  size_t bB2 = (size_t)((n0 + 2) * 16) * 32 + lane;
  size_t bB3 = (size_t)((n0 + 3) * 16) * 32 + lane;
  v8f acc0 = {0.f,0.f,0.f,0.f,0.f,0.f,0.f,0.f};
  v8f acc1 = acc0, acc2 = acc0, acc3 = acc0;
  // distance-1 pipelining: loads for k+1 are issued and *pinned* before wmma(k)
  v16bf a_c = A[aB];
  v16bf b0 = Bw[bB0], b1 = Bw[bB1], b2 = Bw[bB2], b3 = Bw[bB3];
#pragma unroll
  for (int kt = 0; kt < 15; kt++) {
    size_t s = (size_t)(kt + 1) * 32;
    v16bf a_n  = A[aB + s];
    v16bf b0n = Bw[bB0 + s], b1n = Bw[bB1 + s];
    v16bf b2n = Bw[bB2 + s], b3n = Bw[bB3 + s];
    if (kt < 14) __builtin_prefetch((const void*)(A + aB + s + 32), 0, 3);
    sched_fence();
    acc0 = wmma_bf16(a_c, b0, acc0);
    acc1 = wmma_bf16(a_c, b1, acc1);
    acc2 = wmma_bf16(a_c, b2, acc2);
    acc3 = wmma_bf16(a_c, b3, acc3);
    sched_fence();
    a_c = a_n; b0 = b0n; b1 = b1n; b2 = b2n; b3 = b3n;
  }
  acc0 = wmma_bf16(a_c, b0, acc0);
  acc1 = wmma_bf16(a_c, b1, acc1);
  acc2 = wmma_bf16(a_c, b2, acc2);
  acc3 = wmma_bf16(a_c, b3, acc3);
  v8f acc[4] = {acc0, acc1, acc2, acc3};

  int n = lane & 15, h2 = lane >> 4;
  int c0base = n0 * 16;                  // 64-aligned region base
  float* myl = lds + wv * 1024;
  int mrow = lane & 15, kh = lane >> 4;

  if (c0base < 512) {                    // ---- q region
    int bb = mt >> 7, qt = mt & 127, hh = c0base >> 6;
#pragma unroll
    for (int j = 0; j < 4; j++) {
      int gc = c0base + j * 16 + n;
      float bias = mlp_bias[gc];
#pragma unroll
      for (int i = 0; i < 8; i++)
        myl[(h2 * 8 + i) * 64 + j * 16 + n] = (acc[j][i] + bias) * QSCALE;
    }
    lds_fence();
    unsigned short* qb = qfrag + (size_t)(((bb * 8 + hh) * 128 + qt) * 2) * 32 * 16;
#pragma unroll
    for (int kt2 = 0; kt2 < 2; kt2++) {
      const float* base = myl + mrow * 64 + kt2 * 32 + 8 * kh;
      pack16_store(base, base + 16, qb + (size_t)(kt2 * 32 + lane) * 16);
    }
  } else if (c0base < 1536) {            // ---- kv region (f32 for LN2)
#pragma unroll
    for (int j = 0; j < 4; j++) {
      int gc = c0base + j * 16 + n;
      float bias = mlp_bias[gc];
      int cc = gc - 512;
#pragma unroll
      for (int i = 0; i < 8; i++) {
        int m = mt * 16 + h2 * 8 + i;
        bufKV[(size_t)m * 1024 + cc] = acc[j][i] + bias;
      }
    }
  } else {                               // ---- mlp region: GELU -> cat frags
    int ktb = (c0base - 1536) >> 5;
#pragma unroll
    for (int j = 0; j < 4; j++) {
      int gc = c0base + j * 16 + n;
      float bias = (gc < 2048) ? mlp_bias[gc] : 0.f;
#pragma unroll
      for (int i = 0; i < 8; i++) {
        float v = acc[j][i] + bias;
        v = 0.5f * v * (1.f + erff(v * 0.70710678118f));  // exact GELU
        myl[(h2 * 8 + i) * 64 + j * 16 + n] = v;
      }
    }
    lds_fence();
#pragma unroll
    for (int kt2 = 0; kt2 < 2; kt2++) {
      unsigned short* cb = catfrag + (size_t)((mt * 80 + ktb + kt2) * 32) * 16;
      const float* base = myl + mrow * 64 + kt2 * 32 + 8 * kh;
      pack16_store(base, base + 16, cb + (size_t)lane * 16);
    }
  }
}

// ============================================================== LN2 + K/V pack
// one wave per token: layernorm over 1024, pack K (B-frag for QK^T) and V.
__global__ __launch_bounds__(256) void ln2_pack_kernel(
    const float* __restrict__ kv, const float* __restrict__ w,
    const float* __restrict__ b,
    unsigned short* __restrict__ kfrag, unsigned short* __restrict__ vfrag) {
  int m    = (blockIdx.x * blockDim.x + threadIdx.x) >> 5;
  int lane = threadIdx.x & 31;
  const float* row = kv + (size_t)m * 1024 + lane * 32;      // 32 contiguous/lane
  float vals[32], s = 0.f, s2 = 0.f;
#pragma unroll
  for (int t4 = 0; t4 < 8; t4++) {
    float4 r = *(const float4*)(row + t4 * 4);
    vals[t4 * 4 + 0] = r.x; vals[t4 * 4 + 1] = r.y;
    vals[t4 * 4 + 2] = r.z; vals[t4 * 4 + 3] = r.w;
    s += r.x + r.y + r.z + r.w;
    s2 += r.x * r.x + r.y * r.y + r.z * r.z + r.w * r.w;
  }
#pragma unroll
  for (int o = 16; o >= 1; o >>= 1) {
    s  += __shfl_xor(s,  o, 32);
    s2 += __shfl_xor(s2, o, 32);
  }
  float mean = s * (1.f / 1024.f);
  float rstd = rsqrtf(s2 * (1.f / 1024.f) - mean * mean + 1e-5f);
  int bb = m >> 11, nb = m & 2047;
#pragma unroll
  for (int t = 0; t < 32; t++) {
    int c = lane * 32 + t;
    float v = (vals[t] - mean) * rstd * w[c] + b[c];
    if (c < 512) {                       // K: GEMM-K dim = head dim
      int d = c, hh = d >> 6, dl = d & 63, kt = dl >> 5, kl = dl & 31;
      int khh = (kl >> 3) & 1, e = (kl & 7) + ((kl >> 4) << 3);
      int lt = (nb & 15) + 16 * khh;
      kfrag[(size_t)((((bb * 8 + hh) * 2 + kt) * 128 + (nb >> 4)) * 32 + lt) * 16 + e] = f2bf(v);
    } else {                             // V: GEMM-K dim = key index
      int d = c - 512, hh = d >> 6, dl = d & 63, ntd = dl >> 4, nc = dl & 15;
      int kl = nb & 31;
      int khh = (kl >> 3) & 1, e = (kl & 7) + ((kl >> 4) << 3);
      int lt = nc + 16 * khh;
      vfrag[(size_t)((((bb * 8 + hh) * 64 + (nb >> 5)) * 4 + ntd) * 32 + lt) * 16 + e] = f2bf(v);
    }
  }
}

// ============================================================== flash attention
// one wave per (b,h,16-query tile); 64 key blocks of 32; online softmax.
// K/V fragment streams pipelined; softmax VALU overlaps next block's loads.
__global__ __launch_bounds__(256) void attn_kernel(
    const unsigned short* __restrict__ qfrag,
    const unsigned short* __restrict__ kfrag,
    const unsigned short* __restrict__ vfrag,
    unsigned short* __restrict__ catfrag) {
  __shared__ float          ldsO[8 * 1024];
  __shared__ unsigned short ldsP[8 * 512];
  int wv = threadIdx.x >> 5, lane = threadIdx.x & 31;
  int w  = blockIdx.x * 8 + wv;
  int bh = w >> 7, qt = w & 127;
  const v16bf* Q  = (const v16bf*)qfrag;
  const v16bf* Kf = (const v16bf*)kfrag;
  const v16bf* Vf = (const v16bf*)vfrag;
  v16bf qf0 = Q[(size_t)((bh * 128 + qt) * 2 + 0) * 32 + lane];
  v16bf qf1 = Q[(size_t)((bh * 128 + qt) * 2 + 1) * 32 + lane];
  v8f o0 = {0.f,0.f,0.f,0.f,0.f,0.f,0.f,0.f}, o1 = o0, o2 = o0, o3 = o0;
  float rmax[8], rsum[8];
#pragma unroll
  for (int i = 0; i < 8; i++) { rmax[i] = -INFINITY; rsum[i] = 0.f; }
  float* myO = ldsO + wv * 1024;
  unsigned short* myP = ldsP + wv * 512;
  int n = lane & 15, h2 = lane >> 4;
  int mrow = lane & 15, kh = lane >> 4;

  size_t k0B = (size_t)((bh * 2 + 0) * 128) * 32 + lane;   // kt=0 stream
  size_t k1B = (size_t)((bh * 2 + 1) * 128) * 32 + lane;   // kt=1 stream
  size_t vB  = (size_t)(bh * 64) * 4 * 32 + lane;          // V stream

  // current key-block fragments (kb = 0)
  v16bf kc00 = Kf[k0B + 0 * 32], kc10 = Kf[k1B + 0 * 32];  // s0 column tile
  v16bf kc01 = Kf[k0B + 1 * 32], kc11 = Kf[k1B + 1 * 32];  // s1 column tile
  v16bf vc0 = Vf[vB + 0 * 32], vc1 = Vf[vB + 1 * 32];
  v16bf vc2 = Vf[vB + 2 * 32], vc3 = Vf[vB + 3 * 32];

  for (int kb = 0; kb < 64; kb++) {
    // issue next key block's loads; pin them above this block's math
    v16bf kn00 = kc00, kn10 = kc10, kn01 = kc01, kn11 = kc11;
    v16bf vn0 = vc0, vn1 = vc1, vn2 = vc2, vn3 = vc3;
    if (kb < 63) {
      size_t ks = (size_t)(2 * kb + 2) * 32;
      kn00 = Kf[k0B + ks];          kn10 = Kf[k1B + ks];
      kn01 = Kf[k0B + ks + 32];     kn11 = Kf[k1B + ks + 32];
      size_t vs = (size_t)((kb + 1) * 4) * 32;
      vn0 = Vf[vB + vs + 0 * 32];   vn1 = Vf[vB + vs + 1 * 32];
      vn2 = Vf[vB + vs + 2 * 32];   vn3 = Vf[vB + vs + 3 * 32];
      __builtin_prefetch((const void*)(Kf + k0B + ks + 64), 0, 3);
    }
    sched_fence();
    v8f z = {0.f,0.f,0.f,0.f,0.f,0.f,0.f,0.f};
    v8f s0 = wmma_bf16(qf0, kc00, z);
    s0     = wmma_bf16(qf1, kc10, s0);
    v8f s1 = wmma_bf16(qf0, kc01, z);
    s1     = wmma_bf16(qf1, kc11, s1);
#pragma unroll
    for (int i = 0; i < 8; i++) {
      float mx = fmaxf(s0[i], s1[i]);
      mx = fmaxf(mx, __shfl_xor(mx, 1, 32));
      mx = fmaxf(mx, __shfl_xor(mx, 2, 32));
      mx = fmaxf(mx, __shfl_xor(mx, 4, 32));
      mx = fmaxf(mx, __shfl_xor(mx, 8, 32));
      float nm   = fmaxf(rmax[i], mx);
      float corr = __expf(rmax[i] - nm);
      rmax[i] = nm;
      float p0 = __expf(s0[i] - nm);
      float p1 = __expf(s1[i] - nm);
      float ps = p0 + p1;
      ps += __shfl_xor(ps, 1, 32);
      ps += __shfl_xor(ps, 2, 32);
      ps += __shfl_xor(ps, 4, 32);
      ps += __shfl_xor(ps, 8, 32);
      rsum[i] = rsum[i] * corr + ps;
      o0[i] *= corr; o1[i] *= corr; o2[i] *= corr; o3[i] *= corr;
      myP[(h2 * 8 + i) * 32 + n]      = f2bf(p0);
      myP[(h2 * 8 + i) * 32 + 16 + n] = f2bf(p1);
    }
    lds_fence();
    union { uint4 q[2]; v16bf v; } pu;                 // P: D-layout -> A-layout
    pu.q[0] = *(const uint4*)(myP + mrow * 32 + kh * 8);
    pu.q[1] = *(const uint4*)(myP + mrow * 32 + 16 + kh * 8);
    o0 = wmma_bf16(pu.v, vc0, o0);
    o1 = wmma_bf16(pu.v, vc1, o1);
    o2 = wmma_bf16(pu.v, vc2, o2);
    o3 = wmma_bf16(pu.v, vc3, o3);
    sched_fence();
    kc00 = kn00; kc10 = kn10; kc01 = kn01; kc11 = kn11;
    vc0 = vn0; vc1 = vn1; vc2 = vn2; vc3 = vn3;
  }
  // normalize and deposit into cat fragments (cols 2048 + h*64 ..)
#pragma unroll
  for (int i = 0; i < 8; i++) {
    float inv = 1.f / rsum[i];
    myO[(h2 * 8 + i) * 64 + 0 * 16 + n] = o0[i] * inv;
    myO[(h2 * 8 + i) * 64 + 1 * 16 + n] = o1[i] * inv;
    myO[(h2 * 8 + i) * 64 + 2 * 16 + n] = o2[i] * inv;
    myO[(h2 * 8 + i) * 64 + 3 * 16 + n] = o3[i] * inv;
  }
  lds_fence();
  int bb = bh >> 3, hh = bh & 7;
  int mt = bb * 128 + qt;
#pragma unroll
  for (int kt2 = 0; kt2 < 2; kt2++) {
    int kt = 64 + hh * 2 + kt2;
    unsigned short* cb = catfrag + (size_t)((mt * 80 + kt) * 32) * 16;
    const float* base = myO + mrow * 64 + kt2 * 32 + 8 * kh;
    pack16_store(base, base + 16, cb + (size_t)lane * 16);
  }
}

// ============================================================== contract GEMM
// wave computes paired column tiles (nt, nt+32) so the final combine
// out = attn_o + probs*mlp_o is fused with bias, output mask, dual residual.
__global__ __launch_bounds__(256) void gemm_contract_kernel(
    const unsigned short* __restrict__ catfrag,
    const unsigned short* __restrict__ wconf,
    const float* __restrict__ cbias, const float* __restrict__ x,
    const int* __restrict__ emask, const float* __restrict__ probs,
    float* __restrict__ out) {
  int wv = threadIdx.x >> 5, lane = threadIdx.x & 31;
  int w  = blockIdx.x * 8 + wv;
  int mt = w >> 5, nt = w & 31;
  const v16bf* A  = (const v16bf*)catfrag;
  const v16bf* Bw = (const v16bf*)wconf;
  size_t aB  = (size_t)(mt * 80) * 32 + lane;
  size_t b0B = (size_t)(nt * 80) * 32 + lane;
  size_t b1B = (size_t)((nt + 32) * 80) * 32 + lane;
  v8f clo = {0.f,0.f,0.f,0.f,0.f,0.f,0.f,0.f}, chi = clo;
  v16bf a_c = A[aB], l_c = Bw[b0B], h_c = Bw[b1B];
#pragma unroll 8
  for (int kt = 0; kt < 79; kt++) {
    size_t s = (size_t)(kt + 1) * 32;
    v16bf a_n = A[aB + s];
    v16bf l_n = Bw[b0B + s];
    v16bf h_n = Bw[b1B + s];
    __builtin_prefetch((const void*)(A + aB + s + 32), 0, 3);
    sched_fence();
    clo = wmma_bf16(a_c, l_c, clo);
    chi = wmma_bf16(a_c, h_c, chi);
    sched_fence();
    a_c = a_n; l_c = l_n; h_c = h_n;
  }
  clo = wmma_bf16(a_c, l_c, clo);
  chi = wmma_bf16(a_c, h_c, chi);

  int n = lane & 15, h2 = lane >> 4;
  int col = nt * 16 + n;                           // 0..511
  float blo = cbias[col], bhi = cbias[col + 512];
#pragma unroll
  for (int i = 0; i < 8; i++) {
    int m = mt * 16 + h2 * 8 + i;
    int otok = 1024 >> emask[m];
    float r  = x[(size_t)m * DIMD + col];
    float lo = clo[i] + blo; if (col       >= otok) lo = 0.f; lo += r;
    float hi = chi[i] + bhi; if (col + 512 >= otok) hi = 0.f; hi += r;
    out[(size_t)m * DIMD + col] = hi + probs[m] * lo;
  }
}

// ============================================================== launch
extern "C" void kernel_launch(void* const* d_in, const int* in_sizes, int n_in,
                              void* d_out, int out_size, void* d_ws, size_t ws_size,
                              hipStream_t stream) {
  (void)in_sizes; (void)n_in; (void)out_size; (void)ws_size;
  const float* x      = (const float*)d_in[0];
  const int*   emask  = (const int*)  d_in[1];
  const float* probs  = (const float*)d_in[2];
  const float* wexp   = (const float*)d_in[3];
  const float* mlpb   = (const float*)d_in[4];
  const float* wcon   = (const float*)d_in[5];
  const float* cbias  = (const float*)d_in[6];
  const float* n1w    = (const float*)d_in[7];
  const float* n1b    = (const float*)d_in[8];
  const float* n2w    = (const float*)d_in[9];
  const float* n2b    = (const float*)d_in[10];
  float* out = (float*)d_out;

  char* ws = (char*)d_ws;
  size_t off = 0;
  auto take = [&](size_t bytes) { char* p = ws + off; off += (bytes + 255) & ~(size_t)255; return p; };
  unsigned short* hfrag   = (unsigned short*)take((size_t)MTOT * DIMD * 2);
  unsigned short* wexpf   = (unsigned short*)take((size_t)EXPN * DIMD * 2);
  unsigned short* wconf   = (unsigned short*)take((size_t)COND * CATD * 2);
  unsigned short* qfragp  = (unsigned short*)take((size_t)MTOT * DIMD * 2);
  float*          bufKV   = (float*)         take((size_t)MTOT * 1024 * 4);
  unsigned short* kfragp  = (unsigned short*)take((size_t)MTOT * DIMD * 2);
  unsigned short* vfragp  = (unsigned short*)take((size_t)MTOT * DIMD * 2);
  unsigned short* catfrag = (unsigned short*)take((size_t)MTOT * CATD * 2);

  // 1) weight packing into B-fragment layout
  pack_b_kernel<<<(224 * 16 * 32) / 256, 256, 0, stream>>>(wexp, wexpf, 224, 16, DIMD);
  pack_b_kernel<<<(64 * 80 * 32) / 256, 256, 0, stream>>>(wcon, wconf, 64, 80, CATD);
  // 2) LN1 + mask + A-frag pack (1 wave / token)
  ln1_pack_kernel<<<MTOT / 8, 256, 0, stream>>>(x, emask, n1w, n1b, hfrag);
  // 3) expand GEMM (1024 row-tiles x 56 groups of 4 col-tiles = 57344 waves)
  gemm_expand_kernel<<<(1024 * 56) / 8, 256, 0, stream>>>(
      hfrag, wexpf, mlpb, bufKV, qfragp, catfrag);
  // 4) LN2 + K/V fragment pack (1 wave / token)
  ln2_pack_kernel<<<MTOT / 8, 256, 0, stream>>>(bufKV, n2w, n2b, kfragp, vfragp);
  // 5) flash attention (8 b * 8 h * 128 q-tiles = 8192 waves)
  attn_kernel<<<8192 / 8, 256, 0, stream>>>(qfragp, kfragp, vfragp, catfrag);
  // 6) contract GEMM + fused mask/bias/residual/combine (1024 x 32 waves)
  gemm_contract_kernel<<<(1024 * 32) / 8, 256, 0, stream>>>(
      catfrag, wconf, cbias, x, emask, probs, out);
}
